// CosformerAttention_49263274885960
// MI455X (gfx1250) — matrix operand order repaired and checked by reading, not compiled, not run on hardware
//
#include <hip/hip_runtime.h>
#include <hip/hip_bf16.h>

typedef __attribute__((ext_vector_type(16))) _Float16 v16h;
typedef __attribute__((ext_vector_type(8)))  _Float16 v8h;
typedef __attribute__((ext_vector_type(8)))  float    v8f;
typedef __attribute__((ext_vector_type(4)))  unsigned int u32x4;
typedef __attribute__((ext_vector_type(8)))  int      i32x8;
typedef __attribute__((ext_vector_type(4)))  int      i32x4;

#define L_  2048
#define N_  4
#define E_  1024
#define H_  16
#define D_  64
#define NH_ 64
#define D2_ 128
#define R_  8192   // L*N rows

#define PANEL_LDH 136    // 128 cols + 8-half pad -> 272B row stride, conflict-free

#if __has_builtin(__builtin_amdgcn_tensor_load_to_lds) && __has_builtin(__builtin_amdgcn_s_wait_tensorcnt)
#define HAVE_TDM 1
#else
#define HAVE_TDM 0
#endif

// ---------------------------------------------------------------- fragments
__device__ __forceinline__ v16h cat8(v8h lo, v8h hi) {
  return __builtin_shufflevector(lo, hi, 0,1,2,3,4,5,6,7,8,9,10,11,12,13,14,15);
}

// A fragment (16x32 f16) from row-major matrix, row stride `ld` halves.
// ISA 7.12.2: lane&15 = M; lanes>=16 shift K by 8; halves 0..7 cover
// K = kb + 8*(lane>=16) + 0..7, halves 8..15 the same +16.
__device__ __forceinline__ v16h load_a_frag(const _Float16* __restrict__ base,
                                            int ld, int m, int kb, int lane) {
  int kgrp8 = (lane >> 4) << 3;
  const _Float16* p = base + (size_t)m * ld + kb + kgrp8;
  v8h lo = *(const v8h*)p;
  v8h hi = *(const v8h*)(p + 16);
  return cat8(lo, hi);
}

// B fragment (32x16 f16) from row-major (K x N), row stride `ld` halves.
// lane&15 = N; halves 0..7 -> K = kb + 8*(lane>=16) + 0..7, halves 8..15 the same +16.
__device__ __forceinline__ v16h load_b_frag(const _Float16* __restrict__ base,
                                            int ld, int kb, int n0, int lane) {
  int k = kb + ((lane >> 4) << 3);
  const _Float16* p = base + (size_t)k * ld + n0 + (lane & 15);
  v16h r;
#pragma unroll
  for (int s = 0; s < 8; ++s) {
    r[s]     = p[(size_t)s * ld];
    r[8 + s] = p[(size_t)(16 + s) * ld];
  }
  return r;
}

// Same, from the padded LDS panel.
__device__ __forceinline__ v16h load_b_frag_lds(const _Float16* p0, int kb, int n0, int lane) {
  int k = kb + ((lane >> 4) << 3);
  const _Float16* p = p0 + k * PANEL_LDH + n0 + (lane & 15);
  v16h r;
#pragma unroll
  for (int s = 0; s < 8; ++s) {
    r[s]     = p[s * PANEL_LDH];
    r[8 + s] = p[(16 + s) * PANEL_LDH];
  }
  return r;
}

__device__ __forceinline__ v8f zero8() {
  v8f z = {0.f,0.f,0.f,0.f,0.f,0.f,0.f,0.f};
  return z;
}

// ------------------------------------------------- PHM weight materialization
__global__ void phm_weight(const float* __restrict__ A, const float* __restrict__ S,
                           _Float16* __restrict__ W) {
  int idx = blockIdx.x * blockDim.x + threadIdx.x;      // 0 .. 1M-1
  int row = idx >> 10, col = idx & 1023;
  int i = row >> 8, kk = row & 255;
  int j = col >> 8, ll = col & 255;
  float acc = 0.f;
#pragma unroll
  for (int n = 0; n < 4; ++n)
    acc += A[n * 16 + i * 4 + j] * S[(n * 256 + kk) * 256 + ll];
  W[idx] = (_Float16)acc;
}

// ---------------------------------------------------------------- layernorm
__global__ void layernorm(const float* __restrict__ x, const float* __restrict__ g,
                          const float* __restrict__ b, float* __restrict__ y32,
                          _Float16* __restrict__ y16, _Float16* __restrict__ x16) {
  __shared__ float s1[256], s2[256];
  int row = blockIdx.x;
  const float* xr = x + (size_t)row * E_;
  float v[4]; float sum = 0.f, sq = 0.f;
#pragma unroll
  for (int t = 0; t < 4; ++t) {
    float val = xr[threadIdx.x + 256 * t];
    v[t] = val; sum += val; sq += val * val;
  }
  s1[threadIdx.x] = sum; s2[threadIdx.x] = sq; __syncthreads();
  for (int off = 128; off > 0; off >>= 1) {
    if (threadIdx.x < off) {
      s1[threadIdx.x] += s1[threadIdx.x + off];
      s2[threadIdx.x] += s2[threadIdx.x + off];
    }
    __syncthreads();
  }
  float mean = s1[0] * (1.f / E_);
  float var  = s2[0] * (1.f / E_) - mean * mean;
  float rstd = rsqrtf(var + 1e-5f);
#pragma unroll
  for (int t = 0; t < 4; ++t) {
    int c = threadIdx.x + 256 * t;
    float val = (v[t] - mean) * rstd * g[c] + b[c];
    if (y32) y32[(size_t)row * E_ + c] = val;
    y16[(size_t)row * E_ + c] = (_Float16)val;
    if (x16) x16[(size_t)row * E_ + c] = (_Float16)v[t];
  }
}

// ------------------------------------------------------- main PHM GEMMs
// Y(8192x1024) = X @ W + bias. Block: 256 rows x 128 cols, 8 waves, each wave
// a 32x128 strip (16 accumulators). W panel (1024x128 f16, padded) staged in
// LDS by the Tensor Data Mover once per block.
// MODE 0: relu, scatter q_ = [hh][l][ d*sin | d*cos ]         (f16)
// MODE 1: relu, scatter k_T = [hh][dd][l] with sin/cos        (f16)
// MODE 2: scatter v = [hh][l][d]                              (f16)
// MODE 3: + residual(f32), write f32 row-major (final output)
template <int MODE>
__global__ __launch_bounds__(256)
void gemm_phm(const _Float16* __restrict__ X, const _Float16* __restrict__ W,
              const float* __restrict__ bias, _Float16* __restrict__ out16,
              const float* __restrict__ resid, float* __restrict__ out32) {
  __shared__ _Float16 panel[E_ * PANEL_LDH];            // 272 KB of the 320 KB WGP LDS
  int wave = threadIdx.x >> 5, lane = threadIdx.x & 31;
  int rowbase = blockIdx.x * 256 + wave * 32;
  int colbase = blockIdx.y * 128;

#if HAVE_TDM
  if (wave == 0) {
    unsigned lds_off = (unsigned)(size_t)(&panel[0]);   // low 32 bits = LDS offset
    unsigned long long ga = (unsigned long long)(size_t)(W + colbase);
    // D# group0: count=1 | lds_addr | global_addr[56:0] | type=2
    u32x4 g0;
    g0[0] = 1u;
    g0[1] = lds_off;
    g0[2] = (unsigned)ga;
    g0[3] = ((unsigned)(ga >> 32) & 0x01FFFFFFu) | (2u << 30);
    // D# group1: data_size=2B, pad every 64 DW by 4 DW, dims/strides
    i32x8 g1;
    g1[0] = (1 << 16) | (1 << 20) | (5 << 22) | (3 << 25);
    g1[1] = (int)((E_ & 0xFFFF) << 16);                        // tensor_dim0 lo16
    g1[2] = (int)(((E_ >> 16) & 0xFFFF) | ((E_ & 0xFFFF) << 16)); // dim0 hi | dim1 lo
    g1[3] = (int)(((E_ >> 16) & 0xFFFF) | (128 << 16));        // dim1 hi | tile_dim0=128
    g1[4] = (int)E_;                                           // tile_dim1=1024, tile_dim2=0
    g1[5] = (int)E_;                                           // tensor_dim0_stride lo32
    g1[6] = 0;
    g1[7] = 0;
    i32x4 gz = {0, 0, 0, 0};
#if defined(__clang_major__) && (__clang_major__ >= 23)
    i32x8 g4 = {0, 0, 0, 0, 0, 0, 0, 0};
    __builtin_amdgcn_tensor_load_to_lds(g0, g1, gz, gz, g4, 0);
#else
    __builtin_amdgcn_tensor_load_to_lds(g0, g1, gz, gz, 0);
#endif
    __builtin_amdgcn_s_wait_tensorcnt(0);
  }
  __syncthreads();
#else
  // Fallback: cooperative copy with padding.
  for (int u = threadIdx.x; u < E_ * 16; u += 256) {
    int r = u >> 4, s = u & 15;
    *(v8h*)&panel[r * PANEL_LDH + s * 8] =
        *(const v8h*)(W + (size_t)r * E_ + colbase + s * 8);
  }
  __syncthreads();
#endif

  v8f acc[2][8];
#pragma unroll
  for (int g = 0; g < 2; ++g)
#pragma unroll
    for (int t = 0; t < 8; ++t) acc[g][t] = zero8();

  for (int kb = 0; kb < E_; kb += 32) {
    v16h a0 = load_a_frag(X, E_, rowbase +      (lane & 15), kb, lane);
    v16h a1 = load_a_frag(X, E_, rowbase + 16 + (lane & 15), kb, lane);
#pragma unroll
    for (int t = 0; t < 8; ++t) {
      v16h bf = load_b_frag_lds(panel, kb, 16 * t, lane);
      acc[0][t] = __builtin_amdgcn_wmma_f32_16x16x32_f16(
          false, a0, false, bf, (short)0, acc[0][t], false, false);
      acc[1][t] = __builtin_amdgcn_wmma_f32_16x16x32_f16(
          false, a1, false, bf, (short)0, acc[1][t], false, false);
    }
  }

  int mbase = (lane >> 4) << 3;
  int ncol  = lane & 15;
#pragma unroll
  for (int g = 0; g < 2; ++g) {
#pragma unroll
    for (int t = 0; t < 8; ++t) {
      int col = colbase + 16 * t + ncol;
#pragma unroll
      for (int j = 0; j < 8; ++j) {
        int row = rowbase + 16 * g + mbase + j;
        float val = acc[g][t][j] + bias[col];
        if (MODE == 0 || MODE == 1) {
          val = fmaxf(val, 0.f);
          int l = row >> 2, n = row & 3;        // row = l*N + n
          int h = col >> 6, d = col & 63;
          int hh = n * 16 + h;
          float ang = (float)(l + 1) * (1.5707963267948966f / (float)L_);
          float sv = __sinf(ang), cv = __cosf(ang);
          if (MODE == 0) {
            size_t o = ((size_t)hh * L_ + l) * D2_ + d;
            out16[o]      = (_Float16)(val * sv);
            out16[o + 64] = (_Float16)(val * cv);
          } else {
            size_t o = ((size_t)hh * D2_ + d) * L_ + l;
            out16[o]                   = (_Float16)(val * sv);
            out16[o + (size_t)64 * L_] = (_Float16)(val * cv);
          }
        } else if (MODE == 2) {
          int l = row >> 2, n = row & 3;
          int h = col >> 6, d = col & 63;
          int hh = n * 16 + h;
          out16[((size_t)hh * L_ + l) * D_ + d] = (_Float16)val;
        } else {
          size_t o = (size_t)row * E_ + col;
          out32[o] = val + resid[o];
        }
      }
    }
  }
}

// ------------------------------------------------------------ ksum (128/head)
__global__ void ksum_kernel(const _Float16* __restrict__ kT, float* __restrict__ ksum) {
  int hh = blockIdx.x, dd = threadIdx.x;                // 128 threads
  const _Float16* p = kT + ((size_t)hh * D2_ + dd) * L_;
  float s = 0.f;
  for (int l = 0; l < L_; l += 8) {
    v8h vv = *(const v8h*)(p + l);
#pragma unroll
    for (int u = 0; u < 8; ++u) s += (float)vv[u];
  }
  ksum[hh * D2_ + dd] = s;
}

// ------------------------------------------------- kv[hh] = k_^T @ v (128x64)
__global__ __launch_bounds__(256)
void kv_kernel(const _Float16* __restrict__ kT, const _Float16* __restrict__ v,
               _Float16* __restrict__ kv) {
  int hh = blockIdx.x;
  int wave = threadIdx.x >> 5, lane = threadIdx.x & 31;
  const _Float16* kb_ = kT + (size_t)hh * D2_ * L_;     // (128 x 2048)
  const _Float16* vb_ = v  + (size_t)hh * L_ * D_;      // (2048 x 64)
  v8f acc[4];
#pragma unroll
  for (int t = 0; t < 4; ++t) acc[t] = zero8();

  int m = wave * 16 + (lane & 15);                      // dd row
  for (int lb = 0; lb < L_; lb += 32) {
    v16h a = load_a_frag(kb_, L_, m, lb, lane);
#pragma unroll
    for (int t = 0; t < 4; ++t) {
      v16h bf = load_b_frag(vb_, D_, lb, 16 * t, lane);
      acc[t] = __builtin_amdgcn_wmma_f32_16x16x32_f16(
          false, a, false, bf, (short)0, acc[t], false, false);
    }
  }
  int mbase = (lane >> 4) << 3, nc = lane & 15;
#pragma unroll
  for (int t = 0; t < 4; ++t)
#pragma unroll
    for (int j = 0; j < 8; ++j)
      kv[(size_t)hh * (D2_ * D_) + (size_t)(wave * 16 + mbase + j) * D_ + 16 * t + nc]
          = (_Float16)acc[t][j];
}

// ------------------------------------------------------------ z = 1/max(q_.ksum)
__global__ void z_kernel(const _Float16* __restrict__ q_, const float* __restrict__ ksum,
                         float* __restrict__ z) {
  int idx = blockIdx.x * blockDim.x + threadIdx.x;      // hh*2048 + l
  int hh = idx >> 11;
  const _Float16* qp = q_ + (size_t)idx * D2_;
  float s = 0.f;
  for (int d = 0; d < D2_; d += 8) {
    v8h qq = *(const v8h*)(qp + d);
#pragma unroll
    for (int u = 0; u < 8; ++u) s += (float)qq[u] * ksum[hh * D2_ + d + u];
  }
  z[idx] = 1.f / fmaxf(s, 1e-6f);
}

// --------------------------------------- attn = (q_ @ kv) * z, += qn (in place)
__global__ __launch_bounds__(256)
void attn_kernel(const _Float16* __restrict__ q_, const _Float16* __restrict__ kv,
                 const float* __restrict__ z, float* __restrict__ attn) {
  int hh = blockIdx.y;
  int wave = threadIdx.x >> 5, lane = threadIdx.x & 31;
  int lbase = blockIdx.x * 128 + wave * 16;
  const _Float16* qb_  = q_ + (size_t)hh * L_ * D2_;    // (2048 x 128)
  const _Float16* kvb_ = kv + (size_t)hh * D2_ * D_;    // (128 x 64)
  v8f acc[4];
#pragma unroll
  for (int t = 0; t < 4; ++t) acc[t] = zero8();

  for (int kb = 0; kb < D2_; kb += 32) {
    v16h a = load_a_frag(qb_, D2_, lbase + (lane & 15), kb, lane);
#pragma unroll
    for (int t = 0; t < 4; ++t) {
      v16h bf = load_b_frag(kvb_, D_, kb, 16 * t, lane);
      acc[t] = __builtin_amdgcn_wmma_f32_16x16x32_f16(
          false, a, false, bf, (short)0, acc[t], false, false);
    }
  }
  int mbase = (lane >> 4) << 3, nc = lane & 15;
  int n = hh >> 4, h = hh & 15;
#pragma unroll
  for (int t = 0; t < 4; ++t)
#pragma unroll
    for (int j = 0; j < 8; ++j) {
      int l = lbase + mbase + j;
      int d = 16 * t + nc;
      float val = acc[t][j] * z[hh * L_ + l];
      size_t o = ((size_t)(l * N_ + n)) * E_ + h * D_ + d;
      attn[o] += val;                                   // attn buffer holds qn
    }
}

// ---------------------------------------------------------------- launcher
extern "C" void kernel_launch(void* const* d_in, const int* in_sizes, int n_in,
                              void* d_out, int out_size, void* d_ws, size_t ws_size,
                              hipStream_t stream) {
  const float* query = (const float*)d_in[0];
  const float* qA = (const float*)d_in[1];  const float* qS = (const float*)d_in[2];
  const float* qb = (const float*)d_in[3];
  const float* kA = (const float*)d_in[4];  const float* kS = (const float*)d_in[5];
  const float* kb = (const float*)d_in[6];
  const float* vA = (const float*)d_in[7];  const float* vS = (const float*)d_in[8];
  const float* vb = (const float*)d_in[9];
  const float* oA = (const float*)d_in[10]; const float* oS = (const float*)d_in[11];
  const float* ob = (const float*)d_in[12];
  const float* g1 = (const float*)d_in[13]; const float* b1 = (const float*)d_in[14];
  const float* g2 = (const float*)d_in[15]; const float* b2 = (const float*)d_in[16];
  float* outF = (float*)d_out;

  char* ws = (char*)d_ws;
  const size_t MB = 1024ull * 1024ull;
  _Float16* Wq   = (_Float16*)(ws + 0 * MB);            // 2 MB each
  _Float16* Wk   = (_Float16*)(ws + 2 * MB);
  _Float16* Wv   = (_Float16*)(ws + 4 * MB);
  _Float16* Wo   = (_Float16*)(ws + 6 * MB);
  float*    qn32 = (float*)   (ws + 8 * MB);            // 32 MB (later: attn, in place)
  _Float16* qn16 = (_Float16*)(ws + 40 * MB);           // 16 MB
  _Float16* x16  = (_Float16*)(ws + 56 * MB);           // 16 MB
  _Float16* qbuf = (_Float16*)(ws + 72 * MB);           // 32 MB q_  (later: out2 f32)
  _Float16* kT   = (_Float16*)(ws + 104 * MB);          // 32 MB
  _Float16* vbuf = (_Float16*)(ws + 136 * MB);          // 16 MB v   (later: out2 f16)
  _Float16* kvb  = (_Float16*)(ws + 152 * MB);          // 1 MB
  float*    ksum = (float*)   (ws + 153 * MB);          // 32 KB
  float*    zbuf = (float*)   (ws + 154 * MB);          // 512 KB
  float*    out2_32 = (float*)qbuf;                     // alias (q_ dead by then)
  _Float16* out2_16 = vbuf;                             // alias (v dead by then)

  // 1) PHM weights (f16)
  phm_weight<<<4096, 256, 0, stream>>>(qA, qS, Wq);
  phm_weight<<<4096, 256, 0, stream>>>(kA, kS, Wk);
  phm_weight<<<4096, 256, 0, stream>>>(vA, vS, Wv);
  phm_weight<<<4096, 256, 0, stream>>>(oA, oS, Wo);

  // 2) LN1 (qn f32 + f16) and f16 copy of raw query
  layernorm<<<R_, 256, 0, stream>>>(query, g1, b1, qn32, qn16, x16);

  // 3) q/k/v projections with fused epilogues
  dim3 gg(R_ / 256, E_ / 128);
  gemm_phm<0><<<gg, 256, 0, stream>>>(qn16, Wq, qb, qbuf, nullptr, nullptr);
  gemm_phm<1><<<gg, 256, 0, stream>>>(x16,  Wk, kb, kT,   nullptr, nullptr);
  gemm_phm<2><<<gg, 256, 0, stream>>>(x16,  Wv, vb, vbuf, nullptr, nullptr);

  // 4) per-head reductions + attention GEMMs
  ksum_kernel<<<NH_, D2_, 0, stream>>>(kT, ksum);
  kv_kernel<<<NH_, 256, 0, stream>>>(kT, vbuf, kvb);
  z_kernel<<<(NH_ * L_) / 256, 256, 0, stream>>>(qbuf, ksum, zbuf);
  attn_kernel<<<dim3(L_ / 128, NH_), 256, 0, stream>>>(qbuf, kvb, zbuf, qn32);

  // 5) LN2 + output projection with residual
  layernorm<<<R_, 256, 0, stream>>>(qn32, g2, b2, out2_32, out2_16, nullptr);
  gemm_phm<3><<<gg, 256, 0, stream>>>(out2_16, Wo, ob, nullptr, out2_32, outF);

  (void)in_sizes; (void)n_in; (void)out_size; (void)ws_size;
}
